// HGAN_45148696215914
// MI455X (gfx1250) — matrix operand order repaired and checked by reading, not compiled, not run on hardware
//
#include <hip/hip_runtime.h>

#define HEADS 4
#define DM 256            // hidden dim
#define HC (HEADS * DM)   // 1024

typedef unsigned int  u32;
typedef unsigned short u16;
typedef __attribute__((ext_vector_type(16))) __bf16 v16bf;
typedef __attribute__((ext_vector_type(8)))  float  v8f;

union BFrag { u32 u[8]; v16bf v; };

__device__ __forceinline__ u16 f2bf(float f) {
  u32 u = __float_as_uint(f);
  u += 0x7FFFu + ((u >> 16) & 1u);   // round-to-nearest-even
  return (u16)(u >> 16);
}
// monotonic float<->uint mapping for atomic max
__device__ __forceinline__ u32 ford(float f) {
  u32 u = __float_as_uint(f);
  return (u & 0x80000000u) ? ~u : (u | 0x80000000u);
}
__device__ __forceinline__ float fuord(u32 u) {
  return (u & 0x80000000u) ? __uint_as_float(u & 0x7FFFFFFFu) : __uint_as_float(~u);
}
__device__ __forceinline__ float leaky(float x) { return x > 0.f ? x : 0.2f * x; }

// A-matrix 16x32 bf16 fragment (ISA 7.12.2): lane = M row (mod 16), kg = lane>>4
__device__ __forceinline__ void load_a(BFrag& f, const u16* base, int lane) {
  const int kg = (lane >> 4) * 8;
#pragma unroll
  for (int v = 0; v < 4; ++v) f.u[v] = *(const u32*)(base + kg + 2 * v);
#pragma unroll
  for (int v = 0; v < 4; ++v) f.u[4 + v] = *(const u32*)(base + 16 + kg + 2 * v);
}
// B-matrix 32x16 bf16 fragment: lane = N col (mod 16); kg = lane>>4, same K order as A
__device__ __forceinline__ void load_b(BFrag& f, const u16* base, int lane) {
  const int kg = (lane >> 4) * 8;
#pragma unroll
  for (int v = 0; v < 4; ++v) f.u[v] = *(const u32*)(base + kg + 2 * v);
#pragma unroll
  for (int v = 0; v < 4; ++v) f.u[4 + v] = *(const u32*)(base + 16 + kg + 2 * v);
}

// ---------------- generic bf16 WMMA GEMM: C[M,N] (+)= A[M,K] @ B[K,N] ----------------
// block = 128 threads (4 waves), 64x64 tile, 32-wide K steps
template<bool BIAS, bool RELU, bool ACCUM>
__global__ __launch_bounds__(128)
void k_gemm(const float* __restrict__ A, const float* __restrict__ B,
            const float* __restrict__ bias, float* __restrict__ C,
            int M, int N, int K, int lda, int ldb, int ldc)
{
  __shared__ u16 As[64 * 34];   // [m][k], pad 2
  __shared__ u16 Bs[64 * 34];   // [n][k], pad 2
  const int tid = threadIdx.x, w = tid >> 5, lane = tid & 31;
  const int m0 = blockIdx.y * 64, n0 = blockIdx.x * 64;
  v8f acc[4] = {};
  for (int kt = 0; kt < K; kt += 32) {
    {
      int r = tid >> 1, c0 = (tid & 1) * 16;
      int gm = m0 + r;
#pragma unroll
      for (int j = 0; j < 16; ++j) {
        int gk = kt + c0 + j;
        As[r * 34 + c0 + j] = f2bf((gm < M && gk < K) ? A[(size_t)gm * lda + gk] : 0.f);
      }
      int kr = tid >> 2, nc0 = (tid & 3) * 16;
      int gk = kt + kr;
#pragma unroll
      for (int j = 0; j < 16; ++j) {
        int gn = n0 + nc0 + j;
        Bs[(nc0 + j) * 34 + kr] = f2bf((gn < N && gk < K) ? B[(size_t)gk * ldb + gn] : 0.f);
      }
      // prefetch next K tile while WMMAs run on this one (global_prefetch_b8)
      if (kt + 32 < K) {
        if (gm < M) __builtin_prefetch(&A[(size_t)gm * lda + kt + 32 + c0], 0, 1);
        if (n0 + nc0 < N && kt + 32 + kr < K)
          __builtin_prefetch(&B[(size_t)(kt + 32 + kr) * ldb + n0 + nc0], 0, 1);
      }
    }
    __syncthreads();
    BFrag af; load_a(af, &As[(w * 16 + (lane & 15)) * 34], lane);
#pragma unroll
    for (int f = 0; f < 4; ++f) {
      BFrag bf_; load_b(bf_, &Bs[(f * 16 + (lane & 15)) * 34], lane);
      acc[f] = __builtin_amdgcn_wmma_f32_16x16x32_bf16(false, af.v, false, bf_.v,
                                                       (short)0, acc[f], false, false);
    }
    __syncthreads();
  }
#pragma unroll
  for (int f = 0; f < 4; ++f)
#pragma unroll
    for (int r = 0; r < 8; ++r) {
      int m = m0 + w * 16 + r + 8 * (lane >> 4);
      int n = n0 + f * 16 + (lane & 15);
      if (m < M && n < N) {
        float v = acc[f][r];
        if (ACCUM) v += C[(size_t)m * ldc + n];
        if (BIAS)  v += bias[n];
        if (RELU)  v = fmaxf(v, 0.f);
        C[(size_t)m * ldc + n] = v;
      }
    }
}

// ---------------- fused softmax attention: Out = softmax(Q @ KV^T) @ KV ----------------
// Q [M,256], KV [Nk,256]; 64 Q rows per block, KV chunked by 32, online softmax
__global__ __launch_bounds__(128)
void k_flash(const float* __restrict__ Q, const float* __restrict__ KV,
             float* __restrict__ Out, int M, int Nk)
{
  __shared__ u16 Qs[64 * 258];      // [m][c] bf16
  __shared__ u16 KVs[256 * 34];     // phase A: K rowmajor [n*258+c]; phase B: V^T [c*34+k]
  __shared__ u16 Ps[4 * 16 * 34];   // per-wave P transpose scratch
  const int tid = threadIdx.x, w = tid >> 5, lane = tid & 31;
  const int m0 = blockIdx.x * 64;
  {
    int r = tid >> 1, c0 = (tid & 1) * 128;
    const float* src = Q + (size_t)(m0 + r) * DM;
    bool ok = (m0 + r) < M;
    for (int j = 0; j < 128; ++j)
      Qs[r * 258 + c0 + j] = f2bf(ok ? src[c0 + j] : 0.f);
  }
  __syncthreads();
  float mo[8], li[8];
  v8f o[16] = {};
#pragma unroll
  for (int r = 0; r < 8; ++r) { mo[r] = -3.0e38f; li[r] = 0.f; }
  const int kr = tid >> 2, c0 = (tid & 3) * 64;
  for (int j0 = 0; j0 < Nk; j0 += 32) {
    const float* kvrow = KV + (size_t)(j0 + kr) * DM;
    for (int j = 0; j < 64; ++j)                       // stage K rowmajor
      KVs[kr * 258 + c0 + j] = f2bf(kvrow[c0 + j]);
    if (j0 + 32 < Nk) {                                // prefetch next chunk
      __builtin_prefetch(kvrow + (size_t)32 * DM + c0, 0, 1);
      __builtin_prefetch(kvrow + (size_t)32 * DM + c0 + 32, 0, 1);
    }
    __syncthreads();
    v8f s0 = {}, s1 = {};
#pragma unroll
    for (int ks = 0; ks < 8; ++ks) {                   // S = Q tile @ K^T chunk
      BFrag af; load_a(af, &Qs[(w * 16 + (lane & 15)) * 258 + ks * 32], lane);
      BFrag b0; load_b(b0, &KVs[(lane & 15) * 258 + ks * 32], lane);
      BFrag b1; load_b(b1, &KVs[(16 + (lane & 15)) * 258 + ks * 32], lane);
      s0 = __builtin_amdgcn_wmma_f32_16x16x32_bf16(false, af.v, false, b0.v, (short)0, s0, false, false);
      s1 = __builtin_amdgcn_wmma_f32_16x16x32_bf16(false, af.v, false, b1.v, (short)0, s1, false, false);
    }
    __syncthreads();
    for (int j = 0; j < 64; ++j)                       // restage chunk transposed (V^T); L2-hot
      KVs[(c0 + j) * 34 + kr] = f2bf(kvrow[c0 + j]);
#pragma unroll
    for (int r = 0; r < 8; ++r) {                      // online softmax per row
      float mx = fmaxf(s0[r], s1[r]);
#pragma unroll
      for (int d = 1; d < 16; d <<= 1) mx = fmaxf(mx, __shfl_xor(mx, d, 32));
      float mn = fmaxf(mo[r], mx);
      float sc = __expf(mo[r] - mn);
      float p0 = __expf(s0[r] - mn);
      float p1 = __expf(s1[r] - mn);
      float rs = p0 + p1;
#pragma unroll
      for (int d = 1; d < 16; d <<= 1) rs += __shfl_xor(rs, d, 32);
      li[r] = li[r] * sc + rs;
      mo[r] = mn;
#pragma unroll
      for (int f = 0; f < 16; ++f) o[f][r] *= sc;
      int mr = r + 8 * (lane >> 4);
      Ps[w * 544 + mr * 34 + (lane & 15)]      = f2bf(p0);
      Ps[w * 544 + mr * 34 + 16 + (lane & 15)] = f2bf(p1);
    }
    __syncthreads();
    BFrag pa; load_a(pa, &Ps[w * 544 + (lane & 15) * 34], lane);
#pragma unroll
    for (int f = 0; f < 16; ++f) {                     // O += P @ V chunk
      BFrag bv; load_b(bv, &KVs[(f * 16 + (lane & 15)) * 34], lane);
      o[f] = __builtin_amdgcn_wmma_f32_16x16x32_bf16(false, pa.v, false, bv.v, (short)0, o[f], false, false);
    }
    __syncthreads();
  }
#pragma unroll
  for (int f = 0; f < 16; ++f)
#pragma unroll
    for (int r = 0; r < 8; ++r) {
      int m = m0 + w * 16 + r + 8 * (lane >> 4);
      if (m < M) Out[(size_t)m * DM + f * 16 + (lane & 15)] = o[f][r] / (li[r] + 1e-16f);
    }
}

// ---------------- GAT helper kernels ----------------
__global__ void k_gather(const float* __restrict__ emb, const int* __restrict__ idx,
                         float* __restrict__ out, int N) {
  int g = blockIdx.x * blockDim.x + threadIdx.x;
  if (g >= N * DM) return;
  int i = g >> 8, c = g & 255;
  out[g] = emb[(size_t)idx[i] * DM + c];
}

// per-(node,head) wave: s_src = <h, a_src>, s_dst = <h, a_dst>
__global__ void k_scores(const float* __restrict__ h, const float* __restrict__ asrc,
                         const float* __restrict__ adst, float* __restrict__ ssrc,
                         float* __restrict__ sdst, int NH) {
  int wv = (blockIdx.x * blockDim.x + threadIdx.x) >> 5;
  int lane = threadIdx.x & 31;
  if (wv >= NH) return;
  int hh = wv & (HEADS - 1);
  const float* row = h + (size_t)wv * DM;
  const float* as = asrc + hh * DM;
  const float* ad = adst + hh * DM;
  float a = 0.f, b = 0.f;
  for (int c = lane; c < DM; c += 32) { float v = row[c]; a += v * as[c]; b += v * ad[c]; }
#pragma unroll
  for (int d = 1; d < 32; d <<= 1) { a += __shfl_xor(a, d, 32); b += __shfl_xor(b, d, 32); }
  if (lane == 0) { ssrc[wv] = a; sdst[wv] = b; }
}

__global__ void k_edge_max(const float* __restrict__ ssrc, const float* __restrict__ sdst,
                           const int* __restrict__ es, const int* __restrict__ ed,
                           int E, int N, u32* __restrict__ mU) {
  long total = (long)(E + N) * HEADS;
  for (long g = blockIdx.x * (long)blockDim.x + threadIdx.x; g < total;
       g += (long)gridDim.x * blockDim.x) {
    long t = g >> 2; int hh = (int)(g & 3);
    int s = t < E ? es[t] : (int)(t - E);
    int d = t < E ? ed[t] : (int)(t - E);
    float e = leaky(ssrc[s * HEADS + hh] + sdst[d * HEADS + hh]);
    atomicMax(&mU[d * HEADS + hh], ford(e));
  }
}

__global__ void k_edge_sum(const float* __restrict__ ssrc, const float* __restrict__ sdst,
                           const int* __restrict__ es, const int* __restrict__ ed,
                           int E, int N, const u32* __restrict__ mU, float* __restrict__ sS) {
  long total = (long)(E + N) * HEADS;
  for (long g = blockIdx.x * (long)blockDim.x + threadIdx.x; g < total;
       g += (long)gridDim.x * blockDim.x) {
    long t = g >> 2; int hh = (int)(g & 3);
    int s = t < E ? es[t] : (int)(t - E);
    int d = t < E ? ed[t] : (int)(t - E);
    float e = leaky(ssrc[s * HEADS + hh] + sdst[d * HEADS + hh]);
    atomicAdd(&sS[d * HEADS + hh], __expf(e - fuord(mU[d * HEADS + hh])));
  }
}

// one block per edge: agg[dst] += alpha * h[src]  (all 4 heads, 1024 channels)
__global__ __launch_bounds__(256)
void k_aggregate(const float* __restrict__ h, const float* __restrict__ ssrc,
                 const float* __restrict__ sdst, const int* __restrict__ es,
                 const int* __restrict__ ed, int E, int N,
                 const u32* __restrict__ mU, const float* __restrict__ sS,
                 float* __restrict__ agg) {
  long t = blockIdx.x;
  int s = t < E ? es[t] : (int)(t - E);
  int d = t < E ? ed[t] : (int)(t - E);
  float alpha[HEADS];
#pragma unroll
  for (int hh = 0; hh < HEADS; ++hh) {
    float e = leaky(ssrc[s * HEADS + hh] + sdst[d * HEADS + hh]);
    float ex = __expf(e - fuord(mU[d * HEADS + hh]));
    alpha[hh] = ex / (sS[d * HEADS + hh] + 1e-16f);
  }
  const float* hs = h + (size_t)s * HC;
  float* ag = agg + (size_t)d * HC;
  for (int i = threadIdx.x; i < HC; i += 256)
    atomicAdd(&ag[i], alpha[i >> 8] * hs[i]);
}

__global__ void k_finalize(const float* __restrict__ agg, const float* __restrict__ bias,
                           float* __restrict__ out, int N) {
  int g = blockIdx.x * blockDim.x + threadIdx.x;
  if (g >= N * DM) return;
  int n = g >> 8, c = g & 255;
  const float* a = agg + (size_t)n * HC;
  out[g] = 0.25f * (a[c] + a[DM + c] + a[2 * DM + c] + a[3 * DM + c]) + bias[c];
}

// ---------------- host-side orchestration ----------------
static inline dim3 gemm_grid(int M, int N) { return dim3((N + 63) / 64, (M + 63) / 64); }

static void run_gat(const float* feat, int N, int Din,
                    const float* W, const float* asrc, const float* adst, const float* bias,
                    const int* edge, int E,
                    float* h, float* agg, float* ssrc, float* sdst, u32* mU, float* sS,
                    float* out, hipStream_t st)
{
  k_gemm<false, false, false><<<gemm_grid(N, HC), 128, 0, st>>>(
      feat, W, nullptr, h, N, HC, Din, Din, HC, HC);
  int NH = N * HEADS;
  k_scores<<<(NH * 32 + 255) / 256, 256, 0, st>>>(h, asrc, adst, ssrc, sdst, NH);
  hipMemsetAsync(mU, 0, sizeof(u32) * NH, st);
  hipMemsetAsync(sS, 0, sizeof(float) * NH, st);
  hipMemsetAsync(agg, 0, sizeof(float) * (size_t)N * HC, st);
  const int* es = edge; const int* ed = edge + E;
  k_edge_max<<<4096, 256, 0, st>>>(ssrc, sdst, es, ed, E, N, mU);
  k_edge_sum<<<4096, 256, 0, st>>>(ssrc, sdst, es, ed, E, N, mU, sS);
  k_aggregate<<<E + N, 256, 0, st>>>(h, ssrc, sdst, es, ed, E, N, mU, sS, agg);
  k_finalize<<<(N * DM + 255) / 256, 256, 0, st>>>(agg, bias, out, N);
}

extern "C" void kernel_launch(void* const* d_in, const int* in_sizes, int n_in,
                              void* d_out, int out_size, void* d_ws, size_t ws_size,
                              hipStream_t stream)
{
  (void)n_in; (void)out_size; (void)ws_size;
  const float* X    = (const float*)d_in[0];
  const int*   iIdx = (const int*)d_in[1];
  const int*   eIdx = (const int*)d_in[2];
  const int*   dE   = (const int*)d_in[3];
  const int*   iE   = (const int*)d_in[4];
  const int*   eE   = (const int*)d_in[5];
  const float* iEmb = (const float*)d_in[6];
  const float* eEmb = (const float*)d_in[7];
  const float* Wd   = (const float*)d_in[8];
  const float* asd  = (const float*)d_in[9];
  const float* add_ = (const float*)d_in[10];
  const float* bd   = (const float*)d_in[11];
  const float* Wi   = (const float*)d_in[12];
  const float* asi  = (const float*)d_in[13];
  const float* adi  = (const float*)d_in[14];
  const float* bi   = (const float*)d_in[15];
  const float* We   = (const float*)d_in[16];
  const float* ase  = (const float*)d_in[17];
  const float* ade  = (const float*)d_in[18];
  const float* be   = (const float*)d_in[19];
  const float* Wdfc = (const float*)d_in[20];
  const float* bdfc = (const float*)d_in[21];
  const float* Wifc = (const float*)d_in[22];
  const float* bifc = (const float*)d_in[23];
  const float* Wefc = (const float*)d_in[24];
  const float* befc = (const float*)d_in[25];
  const float* Wfc  = (const float*)d_in[26];
  const float* bfc  = (const float*)d_in[27];

  const int Nd = in_sizes[0] / 512;
  const int Ni = in_sizes[1];
  const int Ne = in_sizes[2];
  const int Ed = in_sizes[3] / 2;
  const int Ei = in_sizes[4] / 2;
  const int Ee = in_sizes[5] / 2;
  const int NO = in_sizes[27];     // 32 output classes

  float* ws = (float*)d_ws;
  size_t off = 0;
  auto alloc = [&](size_t n) { float* r = ws + off; off += (n + 63) & ~(size_t)63; return r; };
  float* bh   = alloc((size_t)Nd * HC);       // h buffer (reused per graph)
  float* bagg = alloc((size_t)Nd * HC);       // aggregation buffer (reused)
  float* bss  = alloc((size_t)Nd * HEADS);
  float* bsd  = alloc((size_t)Nd * HEADS);
  float* bm   = alloc((size_t)Nd * HEADS);    // ordered-uint max
  float* bsum = alloc((size_t)Nd * HEADS);
  float* d0   = alloc((size_t)Nd * DM);
  float* dd   = alloc((size_t)Nd * DM);
  float* i0   = alloc((size_t)Ni * DM);
  float* ii   = alloc((size_t)Ni * DM);
  float* e0   = alloc((size_t)Ne * DM);
  float* ee2  = alloc((size_t)Ne * DM);
  float* atti = alloc((size_t)Nd * DM);
  float* atte = alloc((size_t)Nd * DM);
  float* gbuf = alloc((size_t)(Ni > Ne ? Ni : Ne) * DM);

  // dialogue GAT + FC
  run_gat(X, Nd, 512, Wd, asd, add_, bd, dE, Ed, bh, bagg, bss, bsd, (u32*)bm, bsum, d0, stream);
  k_gemm<true, true, false><<<gemm_grid(Nd, DM), 128, 0, stream>>>(
      d0, Wdfc, bdfc, dd, Nd, DM, DM, DM, DM, DM);

  // intention GAT + FC
  k_gather<<<(Ni * DM + 255) / 256, 256, 0, stream>>>(iEmb, iIdx, gbuf, Ni);
  run_gat(gbuf, Ni, DM, Wi, asi, adi, bi, iE, Ei, bh, bagg, bss, bsd, (u32*)bm, bsum, i0, stream);
  k_gemm<true, true, false><<<gemm_grid(Ni, DM), 128, 0, stream>>>(
      i0, Wifc, bifc, ii, Ni, DM, DM, DM, DM, DM);

  // emotion GAT + FC
  k_gather<<<(Ne * DM + 255) / 256, 256, 0, stream>>>(eEmb, eIdx, gbuf, Ne);
  run_gat(gbuf, Ne, DM, We, ase, ade, be, eE, Ee, bh, bagg, bss, bsd, (u32*)bm, bsum, e0, stream);
  k_gemm<true, true, false><<<gemm_grid(Ne, DM), 128, 0, stream>>>(
      e0, Wefc, befc, ee2, Ne, DM, DM, DM, DM, DM);

  // fused cross-attention (softmax(d@i.T)@i, softmax(d@e.T)@e)
  k_flash<<<(Nd + 63) / 64, 128, 0, stream>>>(dd, ii, atti, Nd, Ni);
  k_flash<<<(Nd + 63) / 64, 128, 0, stream>>>(dd, ee2, atte, Nd, Ne);

  // final classifier without materializing concat: out = d@Wfc[0:256] + atti@Wfc[256:512] + atte@Wfc[512:768] + b
  float* out = (float*)d_out;
  k_gemm<true, false, false><<<gemm_grid(Nd, NO), 128, 0, stream>>>(
      dd, Wfc, bfc, out, Nd, NO, DM, DM, NO, NO);
  k_gemm<false, false, true><<<gemm_grid(Nd, NO), 128, 0, stream>>>(
      atti, Wfc + (size_t)DM * NO, nullptr, out, Nd, NO, DM, DM, NO, NO);
  k_gemm<false, false, true><<<gemm_grid(Nd, NO), 128, 0, stream>>>(
      atte, Wfc + (size_t)2 * DM * NO, nullptr, out, Nd, NO, DM, DM, NO, NO);
}
